// MLP_75110388073059
// MI455X (gfx1250) — matrix operand order repaired and checked by reading, not compiled, not run on hardware
//
#include <hip/hip_runtime.h>
#include <hip/hip_bf16.h>

typedef __attribute__((ext_vector_type(16))) _Float16 v16h;
typedef __attribute__((ext_vector_type(8)))  _Float16 h8;
typedef __attribute__((ext_vector_type(8)))  float    v8f;

#define NBINS 32
#define CCH   64
#define EPSV  1e-3f

#define WMMA_F16(a, b, c) \
  __builtin_amdgcn_wmma_f32_16x16x32_f16(false, (a), false, (b), (short)0, (c), false, false)

static __device__ __forceinline__ v16h make16(h8 lo, h8 hi) {
  return __builtin_shufflevector(lo, hi, 0,1,2,3,4,5,6,7,8,9,10,11,12,13,14,15);
}

// A-matrix 16x32 f16 fragment: lanes 0-15 row M=lane, halves 0..15 = K k0..k0+15;
// lanes 16-31 shifted by 16 in K (matches B fragments).
static __device__ __forceinline__ v16h load_a_frag(const _Float16* base, int stride,
                                                   int rt, int kb, int lane) {
  int m = rt * 16 + (lane & 15);
  int g = (lane >> 4) & 1;
  const h8* lo = (const h8*)(base + m * stride + kb * 32 + g * 16);
  const h8* hi = (const h8*)(base + m * stride + kb * 32 + g * 16 + 8);
  return make16(*lo, *hi);
}

// B-matrix 32x16 f16 fragment from a row-major [n][kk] LDS buffer:
// lane L holds column n = nb*16 + L%16, halves = K kb*32 + (L/16)*16 + e.
static __device__ __forceinline__ v16h load_b_frag(const _Float16* base, int stride,
                                                   int kb, int nb, int lane) {
  int n = nb * 16 + (lane & 15);
  int g = (lane >> 4) & 1;
  const h8* lo = (const h8*)(base + n * stride + kb * 32 + g * 16);
  const h8* hi = (const h8*)(base + n * stride + kb * 32 + g * 16 + 8);
  return make16(*lo, *hi);
}

// ---------------------------------------------------------------------------
// Pre-shuffle Wq/Wk/Wv (64x64 f32) into f16 B-fragment order:
// wfrag[m][kb*4+ct][lane][e]  (3 * 8 * 32 * 16 halves)
// ---------------------------------------------------------------------------
__global__ __launch_bounds__(32) void wfrag_kernel(const float* __restrict__ Wq,
                                                   const float* __restrict__ Wk,
                                                   const float* __restrict__ Wv,
                                                   _Float16* __restrict__ wfrag) {
  int m = blockIdx.x >> 3;
  int f = blockIdx.x & 7;
  int kb = f >> 2, ct = f & 3;
  int lane = threadIdx.x;
  const float* W = (m == 0) ? Wq : (m == 1) ? Wk : Wv;
  int n = ct * 16 + (lane & 15);
  int kbase = kb * 32 + ((lane >> 4) & 1) * 16;
  _Float16* dst = wfrag + ((size_t)(m * 8 + f) * 32 + lane) * 16;
#pragma unroll
  for (int e = 0; e < 16; ++e) dst[e] = (_Float16)W[(kbase + e) * CCH + n];
}

// ---------------------------------------------------------------------------
// Pass 1: per-channel sum / sumsq of x = points[:,1:9] @ Wc^T
// ---------------------------------------------------------------------------
__global__ __launch_bounds__(256) void bn_stats_kernel(const float* __restrict__ pts,
                                                       const float* __restrict__ Wc,
                                                       float* __restrict__ bn, int N) {
  __shared__ float feat[64][8];
  int t = threadIdx.x;
  int c = t & 63;
  int pl = t >> 6;  // 0..3
  float w[8];
#pragma unroll
  for (int j = 0; j < 8; ++j) w[j] = Wc[c * 8 + j];
  float s1 = 0.f, s2 = 0.f;
  int chunks = (N + 63) >> 6;
  for (int ch = blockIdx.x; ch < chunks; ch += gridDim.x) {
    int base = ch << 6;
    __syncthreads();
    for (int idx = t; idx < 512; idx += 256) {
      int p = base + (idx >> 3);
      feat[idx >> 3][idx & 7] = (p < N) ? pts[(size_t)p * 9 + 1 + (idx & 7)] : 0.f;
    }
    __syncthreads();
    for (int i = 0; i < 16; ++i) {
      int p = base + i * 4 + pl;
      if (p < N) {
        float x = 0.f;
#pragma unroll
        for (int j = 0; j < 8; ++j) x += feat[i * 4 + pl][j] * w[j];
        s1 += x;
        s2 += x * x;
      }
    }
  }
  atomicAdd(&bn[c], s1);
  atomicAdd(&bn[64 + c], s2);
}

__global__ void bn_finalize_kernel(float* __restrict__ bn, const float* __restrict__ gamma,
                                   const float* __restrict__ beta, int N) {
  int c = threadIdx.x;
  if (c < CCH) {
    float inv_n = 1.f / (float)N;
    float mu = bn[c] * inv_n;
    float var = bn[64 + c] * inv_n - mu * mu;
    float sc = gamma[c] * rsqrtf(var + EPSV);
    bn[128 + c] = sc;                 // scale
    bn[192 + c] = beta[c] - mu * sc;  // bias
  }
}

// ---------------------------------------------------------------------------
// Pass 2: recompute x, BN+ReLU, add sparse_feat[pillar_inv], segment-sum into voxels
// ---------------------------------------------------------------------------
__global__ __launch_bounds__(256) void scatter_kernel(
    const float* __restrict__ pts, const float* __restrict__ Wc, const float* __restrict__ bn,
    const float* __restrict__ sfeat, const int* __restrict__ pinv, const int* __restrict__ vinv,
    float* __restrict__ vsum, float* __restrict__ vcnt, int N) {
  __shared__ float feat[64][8];
  int t = threadIdx.x;
  int c = t & 63;
  int pl = t >> 6;
  float w[8];
#pragma unroll
  for (int j = 0; j < 8; ++j) w[j] = Wc[c * 8 + j];
  float A = bn[128 + c], B = bn[192 + c];
  int chunks = (N + 63) >> 6;
  for (int ch = blockIdx.x; ch < chunks; ch += gridDim.x) {
    int base = ch << 6;
    __syncthreads();
    for (int idx = t; idx < 512; idx += 256) {
      int p = base + (idx >> 3);
      feat[idx >> 3][idx & 7] = (p < N) ? pts[(size_t)p * 9 + 1 + (idx & 7)] : 0.f;
    }
    __syncthreads();
    for (int i = 0; i < 16; ++i) {
      int p = base + i * 4 + pl;
      if (p < N) {
        float x = 0.f;
#pragma unroll
        for (int j = 0; j < 8; ++j) x += feat[i * 4 + pl][j] * w[j];
        x = fmaxf(x * A + B, 0.f);
        int pi = pinv[p], vi = vinv[p];
        float val = x + sfeat[(size_t)pi * CCH + c];
        atomicAdd(&vsum[(size_t)vi * CCH + c], val);
        if (c == 0) atomicAdd(&vcnt[vi], 1.0f);
      }
    }
  }
}

__global__ void occ_rank_kernel(const int* __restrict__ occ, int* __restrict__ occ_rank, int K) {
  int i = blockIdx.x * 256 + threadIdx.x;
  if (i < K) occ_rank[occ[i]] = i;
}

// vfeat = sums / counts (in place); build occ_map[k][bin] = voxel index (or -1)
__global__ void vfeat_kernel(float* __restrict__ vsum, const float* __restrict__ vcnt,
                             const int* __restrict__ occ_rank, const int* __restrict__ pov,
                             const int* __restrict__ binv, int* __restrict__ occ_map, int V) {
  int e = blockIdx.x * 256 + threadIdx.x;
  if (e < V * CCH) {
    int v = e >> 6, c = e & 63;
    vsum[e] = vsum[e] / vcnt[v];
    if (c == 0) {
      int r = occ_rank[pov[v]];
      if (r >= 0) occ_map[r * NBINS + binv[v]] = v;
    }
  }
}

// ---------------------------------------------------------------------------
// Per-pillar attention: one wave32 per occupied pillar, WMMA f16->f32
// ---------------------------------------------------------------------------
__global__ __launch_bounds__(32) void attn_kernel(const float* __restrict__ vfeat,
                                                  const int* __restrict__ occ_map,
                                                  const int* __restrict__ occ_idx,
                                                  const _Float16* __restrict__ wfrag,
                                                  float* __restrict__ out) {
  __shared__ __align__(16) _Float16 s16[NBINS][CCH];   // s, f16
  __shared__ __align__(16) float    s32[NBINS][CCH];   // s, f32 (residual)
  __shared__ __align__(16) _Float16 q16[NBINS][CCH];   // q row-major [bin][ch]
  __shared__ __align__(16) _Float16 k16[NBINS][CCH];   // k row-major [bin][ch] -> B of k^T
  __shared__ __align__(16) _Float16 vT16[CCH][NBINS];  // v transposed [ch][bin] -> B of v
  __shared__ __align__(16) float    sc32[NBINS][NBINS];
  __shared__ __align__(16) _Float16 a16[NBINS][NBINS];
  __shared__ float pool2[CCH][2];

  const int lane = threadIdx.x;
  const int pk = blockIdx.x;
  const int colL = lane & 15;
  const int g = (lane >> 4) & 1;

  // Gather s tile (32 bins x 64 ch); empty bins are zero (reference semantics).
  for (int b = 0; b < NBINS; ++b) {
    int v = occ_map[pk * NBINS + b];  // uniform per b
    for (int c = lane; c < CCH; c += 32) {
      float val = (v >= 0) ? vfeat[(size_t)v * CCH + c] : 0.f;
      s32[b][c] = val;
      s16[b][c] = (_Float16)val;
    }
  }
  __syncthreads();

  // A-fragments of s, reused for q/k/v projections
  v16h sa[2][2];
#pragma unroll
  for (int rt = 0; rt < 2; ++rt)
#pragma unroll
    for (int kb = 0; kb < 2; ++kb) sa[rt][kb] = load_a_frag(&s16[0][0], CCH, rt, kb, lane);

  // q = s@Wq, k = s@Wk, v = s@Wv  (32x64 = 2x4 tiles, K=64 = 2 steps)
#pragma unroll
  for (int m = 0; m < 3; ++m) {
#pragma unroll
    for (int rt = 0; rt < 2; ++rt) {
#pragma unroll
      for (int ct = 0; ct < 4; ++ct) {
        v8f acc = {};
#pragma unroll
        for (int kb = 0; kb < 2; ++kb) {
          const h8* wp = (const h8*)(wfrag + ((size_t)(m * 8 + kb * 4 + ct) * 32 + lane) * 16);
          v16h bfr = make16(wp[0], wp[1]);
          acc = WMMA_F16(sa[rt][kb], bfr, acc);
        }
#pragma unroll
        for (int j = 0; j < 8; ++j) {
          int row = rt * 16 + g * 8 + j;
          int col = ct * 16 + colL;
          if (m == 0)      q16[row][col] = (_Float16)acc[j];
          else if (m == 1) k16[row][col] = (_Float16)acc[j];
          else             vT16[col][row] = (_Float16)acc[j];
        }
      }
    }
  }
  __syncthreads();

  // scores = q @ k^T / 8  (32x32 = 2x2 tiles, K=64)
  v16h qa[2][2];
#pragma unroll
  for (int rt = 0; rt < 2; ++rt)
#pragma unroll
    for (int kb = 0; kb < 2; ++kb) qa[rt][kb] = load_a_frag(&q16[0][0], CCH, rt, kb, lane);
#pragma unroll
  for (int rt = 0; rt < 2; ++rt) {
#pragma unroll
    for (int ct = 0; ct < 2; ++ct) {
      v8f acc = {};
#pragma unroll
      for (int kb = 0; kb < 2; ++kb)
        acc = WMMA_F16(qa[rt][kb], load_b_frag(&k16[0][0], CCH, kb, ct, lane), acc);
#pragma unroll
      for (int j = 0; j < 8; ++j) sc32[rt * 16 + g * 8 + j][ct * 16 + colL] = acc[j] * 0.125f;
    }
  }
  __syncthreads();

  // Softmax: lane = row, 32 cols in LDS
  {
    float mx = -1e30f;
    for (int cc = 0; cc < NBINS; ++cc) mx = fmaxf(mx, sc32[lane][cc]);
    float sum = 0.f;
    for (int cc = 0; cc < NBINS; ++cc) {
      float e = __expf(sc32[lane][cc] - mx);
      sc32[lane][cc] = e;
      sum += e;
    }
    float inv = 1.f / sum;
    for (int cc = 0; cc < NBINS; ++cc) a16[lane][cc] = (_Float16)(sc32[lane][cc] * inv);
  }
  __syncthreads();

  // o = attn @ v + s ; pooled = max over bins
  float cmax[4] = {-1e30f, -1e30f, -1e30f, -1e30f};
#pragma unroll
  for (int rt = 0; rt < 2; ++rt) {
    v16h aa = load_a_frag(&a16[0][0], NBINS, rt, 0, lane);
#pragma unroll
    for (int ct = 0; ct < 4; ++ct) {
      v8f acc = {};
      acc = WMMA_F16(aa, load_b_frag(&vT16[0][0], NBINS, 0, ct, lane), acc);
#pragma unroll
      for (int j = 0; j < 8; ++j) {
        float o = acc[j] + s32[rt * 16 + g * 8 + j][ct * 16 + colL];
        cmax[ct] = fmaxf(cmax[ct], o);
      }
    }
  }
#pragma unroll
  for (int ct = 0; ct < 4; ++ct) pool2[ct * 16 + colL][g] = cmax[ct];
  __syncthreads();

  int p = occ_idx[pk];
  for (int c = lane; c < CCH; c += 32)
    out[(size_t)p * CCH + c] += fmaxf(pool2[c][0], pool2[c][1]);
}

// ---------------------------------------------------------------------------
extern "C" void kernel_launch(void* const* d_in, const int* in_sizes, int n_in,
                              void* d_out, int out_size, void* d_ws, size_t ws_size,
                              hipStream_t stream) {
  const float* pts   = (const float*)d_in[0];
  const float* sfeat = (const float*)d_in[1];
  const float* Wc    = (const float*)d_in[2];
  const float* gamma = (const float*)d_in[3];
  const float* beta  = (const float*)d_in[4];
  const float* Wq    = (const float*)d_in[5];
  const float* Wk    = (const float*)d_in[6];
  const float* Wv    = (const float*)d_in[7];
  const int* pinv = (const int*)d_in[8];
  const int* vinv = (const int*)d_in[9];
  const int* pov  = (const int*)d_in[10];
  const int* binv = (const int*)d_in[11];
  const int* occ  = (const int*)d_in[12];

  int N = in_sizes[8];        // points
  int M = in_sizes[1] / CCH;  // pillars
  int V = in_sizes[10];       // voxels
  int K = in_sizes[12];       // occupied pillars

  char* wp = (char*)d_ws;
  auto carve = [&](size_t bytes) {
    char* r = wp;
    wp += (bytes + 255) & ~(size_t)255;
    return r;
  };
  float*    bn       = (float*)carve(256 * sizeof(float));
  _Float16* wfrag    = (_Float16*)carve(3 * 8 * 32 * 16 * sizeof(_Float16));
  float*    vsum     = (float*)carve((size_t)V * CCH * sizeof(float));
  float*    vcnt     = (float*)carve((size_t)V * sizeof(float));
  int*      occ_rank = (int*)carve((size_t)M * sizeof(int));
  int*      occ_map  = (int*)carve((size_t)K * NBINS * sizeof(int));

  hipMemsetAsync(bn, 0, 256 * sizeof(float), stream);
  hipMemsetAsync(vsum, 0, (size_t)V * CCH * sizeof(float), stream);
  hipMemsetAsync(vcnt, 0, (size_t)V * sizeof(float), stream);
  hipMemsetAsync(occ_rank, 0xFF, (size_t)M * sizeof(int), stream);
  hipMemsetAsync(occ_map, 0xFF, (size_t)K * NBINS * sizeof(int), stream);
  hipMemcpyAsync(d_out, d_in[1], (size_t)M * CCH * sizeof(float),
                 hipMemcpyDeviceToDevice, stream);

  wfrag_kernel<<<24, 32, 0, stream>>>(Wq, Wk, Wv, wfrag);
  bn_stats_kernel<<<512, 256, 0, stream>>>(pts, Wc, bn, N);
  bn_finalize_kernel<<<1, 64, 0, stream>>>(bn, gamma, beta, N);
  scatter_kernel<<<512, 256, 0, stream>>>(pts, Wc, bn, sfeat, pinv, vinv, vsum, vcnt, N);
  if (K > 0) occ_rank_kernel<<<(K + 255) / 256, 256, 0, stream>>>(occ, occ_rank, K);
  vfeat_kernel<<<(V * CCH + 255) / 256, 256, 0, stream>>>(vsum, vcnt, occ_rank, pov, binv,
                                                          occ_map, V);
  if (K > 0)
    attn_kernel<<<K, 32, 0, stream>>>(vsum, occ_map, occ, wfrag, (float*)d_out);
}
